// UNet_NOBlock_83348135346575
// MI455X (gfx1250) — matrix operand order repaired and checked by reading, not compiled, not run on hardware
//
#include <hip/hip_runtime.h>

// ---------------- sizes ----------------
#define B_  2
#define S_  4096
#define NV_ 8
#define D_  256
#define E_  256
#define H_  8
#define HD_ 32
#define HID_ 1024
#define M_  2048            // NV_*D_
#define T_  8192            // B_*S_
#define R_  65536           // T_*NV_

// ---------------- WMMA types ----------------
typedef __attribute__((ext_vector_type(16))) __bf16        bfx16;
typedef __attribute__((ext_vector_type(8)))  float         v8f;
typedef __attribute__((ext_vector_type(4)))  unsigned int  u32x4;

union Frag16 {
    u32x4 q[2];
    unsigned short us[16];
    bfx16 bf;
};

__device__ inline unsigned short f2bf(float f) {
    unsigned int u = __float_as_uint(f);
    u += 0x7FFFu + ((u >> 16) & 1u);          // round-to-nearest-even
    return (unsigned short)(u >> 16);
}
__device__ inline float bf2f(unsigned short h) {
    return __uint_as_float(((unsigned int)h) << 16);
}
__device__ inline float gelu_t(float x) {     // tanh approximation (jax default)
    float u = 0.7978845608028654f * (x + 0.044715f * x * x * x);
    return 0.5f * x * (1.0f + tanhf(u));
}

// A fragment (16x32 bf16) from row-major bf16 matrix, leading dim ld.
// lanes 0-15: row=lane,  K in {kb..kb+7} U {kb+16..kb+23}
// lanes 16-31: row=lane-16, K in {kb+8..kb+15} U {kb+24..kb+31}
__device__ inline bfx16 load_a_frag(const unsigned short* __restrict__ base,
                                    int ld, int rowBase, int kb, int lane) {
    int m = lane & 15, half = lane >> 4;
    const unsigned short* p = base + (size_t)(rowBase + m) * ld + kb + half * 8;
    Frag16 f;
    f.q[0] = *(const u32x4*)(p);
    f.q[1] = *(const u32x4*)(p + 16);
    return f.bf;
}

// B fragment from pre-packed weights: [(kt*NT+nt)*32 + lane][16] contiguous
__device__ inline bfx16 load_b_frag(const unsigned short* __restrict__ packed,
                                    int NT, int kt, int nt, int lane) {
    const unsigned short* p = packed + (((size_t)(kt * NT + nt) * 32 + lane) << 4);
    Frag16 f;
    f.q[0] = *(const u32x4*)(p);
    f.q[1] = *(const u32x4*)(p + 8);
    return f.bf;
}

__device__ inline v8f wmma_bf16(bfx16 a, bfx16 b, v8f c) {
    return __builtin_amdgcn_wmma_f32_16x16x32_bf16(false, a, false, b, (short)0, c, false, false);
}

// ---------------- kernel 1: mod = emb @ w_mod + b_mod (both mods) ----------------
__global__ __launch_bounds__(256)
void mod_kernel(const float* __restrict__ emb,
                const float* __restrict__ w1, const float* __restrict__ bb1,
                const float* __restrict__ w2, const float* __restrict__ bb2,
                float* __restrict__ mod1, float* __restrict__ mod2) {
    int idx = blockIdx.x * 256 + threadIdx.x;     // 2 * B * 2M = 16384
    int sel = idx >> 13;
    int rem = idx & 8191;
    int b   = rem >> 12;
    int j   = rem & 4095;
    const float* w    = sel ? w2  : w1;
    const float* bias = sel ? bb2 : bb1;
    float acc = bias[j];
    const float* e = emb + b * E_;
    for (int k = 0; k < E_; ++k) acc += e[k] * w[(size_t)k * 4096 + j];
    (sel ? mod2 : mod1)[b * 4096 + j] = acc;
}

// ---------------- kernel 2: pack fp32 [K,N] weight into bf16 WMMA-B tiles ----------------
__global__ __launch_bounds__(256)
void pack_kernel(const float* __restrict__ src, unsigned short* __restrict__ dst,
                 int K, int N) {
    int idx = blockIdx.x * 256 + threadIdx.x;
    if (idx >= K * N) return;
    int e  = idx & 15;
    int l  = (idx >> 4) & 31;
    int tt = idx >> 9;
    int NT = N >> 4;
    int nt = tt % NT, kt = tt / NT;
    int kk = kt * 32 + (l >> 4) * 8 + (e & 7) + (e >> 3) * 16;
    int nn = nt * 16 + (l & 15);
    dst[idx] = f2bf(src[(size_t)kk * N + nn]);
}

// ---------------- kernel 3: adaLN over 2048 feats per token -> bf16 ----------------
__global__ __launch_bounds__(256)
void adaln_kernel(const float* __restrict__ xsrc, const float* __restrict__ mod,
                  unsigned short* __restrict__ hout) {
    int t   = blockIdx.x;          // token
    int b   = t / S_;
    int tid = threadIdx.x;
    const float* xp = xsrc + (size_t)t * M_;
    float r[8], sum = 0.f, sq = 0.f;
#pragma unroll
    for (int i = 0; i < 8; ++i) {
        float v = xp[i * 256 + tid];
        r[i] = v; sum += v; sq += v * v;
    }
    __shared__ float s1[256], s2[256];
    s1[tid] = sum; s2[tid] = sq; __syncthreads();
    for (int off = 128; off > 0; off >>= 1) {
        if (tid < off) { s1[tid] += s1[tid + off]; s2[tid] += s2[tid + off]; }
        __syncthreads();
    }
    float mean = s1[0] * (1.f / 2048.f);
    float var  = s2[0] * (1.f / 2048.f) - mean * mean;
    float rstd = rsqrtf(var + 1e-5f);
    const float* mrow = mod + (size_t)b * (2 * M_);
    unsigned short* op = hout + (size_t)t * M_;
#pragma unroll
    for (int i = 0; i < 8; ++i) {
        int j = i * 256 + tid;
        float nv = (r[i] - mean) * rstd;
        op[j] = f2bf(nv * (1.f + mrow[j]) + mrow[M_ + j]);
    }
}

// ---------------- kernel 4: QKV GEMM  h[R,256] @ W[256,256] + b -> bf16 ----------------
__global__ __launch_bounds__(128)
void gemm_qkv_kernel(const unsigned short* __restrict__ h,
                     const unsigned short* __restrict__ wqkv_pack,
                     const float* __restrict__ bq, const float* __restrict__ bk,
                     const float* __restrict__ bv,
                     unsigned short* __restrict__ qkv) {
    int y = blockIdx.y;                                 // 0=q 1=k 2=v
    const unsigned short* wp = wqkv_pack + (size_t)y * (D_ * D_);
    const float* bias = (y == 0) ? bq : ((y == 1) ? bk : bv);
    unsigned short* outp = qkv + (size_t)y * R_ * D_;

    int rowBase = blockIdx.x * 16;
    int wv_ = threadIdx.x >> 5, lane = threadIdx.x & 31;
    v8f acc[4];
#pragma unroll
    for (int j = 0; j < 4; ++j) acc[j] = (v8f){0,0,0,0,0,0,0,0};

    const int NT = D_ / 16;
#pragma unroll
    for (int kt = 0; kt < 8; ++kt) {
        bfx16 a = load_a_frag(h, D_, rowBase, kt * 32, lane);
#pragma unroll
        for (int j = 0; j < 4; ++j) {
            bfx16 b = load_b_frag(wp, NT, kt, wv_ * 4 + j, lane);
            acc[j] = wmma_bf16(a, b, acc[j]);
        }
    }
    int m0 = lane & 15, half = lane >> 4;
#pragma unroll
    for (int j = 0; j < 4; ++j) {
        int col = (wv_ * 4 + j) * 16 + m0;
        float bb = bias[col];
#pragma unroll
        for (int r = 0; r < 8; ++r) {
            int row = rowBase + r + 8 * half;
            outp[(size_t)row * D_ + col] = f2bf(acc[j][r] + bb);
        }
    }
}

// ---------------- kernel 5: attention, one wave per (token, head) ----------------
__global__ __launch_bounds__(256)
void attn_kernel(const unsigned short* __restrict__ qb,
                 const unsigned short* __restrict__ kb,
                 const unsigned short* __restrict__ vb,
                 const unsigned char* __restrict__ mask,
                 unsigned short* __restrict__ ob) {
    int gw   = blockIdx.x * 8 + (threadIdx.x >> 5);
    int lane = threadIdx.x & 31;
    int t = gw >> 3, head = gw & 7;
    int colbase = head * HD_ + lane;
    size_t rb = (size_t)t * NV_ * D_;

    float qv[NV_], kv[NV_], vv[NV_];
    bool  mk[NV_];
#pragma unroll
    for (int n = 0; n < NV_; ++n) {
        size_t idx = rb + (size_t)n * D_ + colbase;
        qv[n] = bf2f(qb[idx]); kv[n] = bf2f(kb[idx]); vv[n] = bf2f(vb[idx]);
        mk[n] = mask[(size_t)t * NV_ + n] != 0;
    }
    const float scale = 0.17677669529663687f;   // 1/sqrt(32)
#pragma unroll
    for (int n = 0; n < NV_; ++n) {
        float sr[NV_];
#pragma unroll
        for (int m = 0; m < NV_; ++m) {
            float p = qv[n] * kv[m];
            p += __shfl_xor(p, 16, 32);
            p += __shfl_xor(p, 8, 32);
            p += __shfl_xor(p, 4, 32);
            p += __shfl_xor(p, 2, 32);
            p += __shfl_xor(p, 1, 32);
            sr[m] = mk[m] ? p * scale : -1e9f;
        }
        float mx = sr[0];
#pragma unroll
        for (int m = 1; m < NV_; ++m) mx = fmaxf(mx, sr[m]);
        float den = 0.f, acc = 0.f;
#pragma unroll
        for (int m = 0; m < NV_; ++m) {
            float e = __expf(sr[m] - mx);
            den += e; acc += e * vv[m];
        }
        ob[rb + (size_t)n * D_ + colbase] = f2bf(acc / den);
    }
}

// ---------------- kernel 6: out-proj GEMM + gamma residual -> x1 in d_out ----------------
__global__ __launch_bounds__(128)
void gemm_oproj_kernel(const unsigned short* __restrict__ o,
                       const unsigned short* __restrict__ wop,
                       const float* __restrict__ bo,
                       const float* __restrict__ gamma,
                       const float* __restrict__ x,
                       float* __restrict__ dout) {
    int rowBase = blockIdx.x * 16;
    int wv_ = threadIdx.x >> 5, lane = threadIdx.x & 31;
    v8f acc[4];
#pragma unroll
    for (int j = 0; j < 4; ++j) acc[j] = (v8f){0,0,0,0,0,0,0,0};
    const int NT = D_ / 16;
#pragma unroll
    for (int kt = 0; kt < 8; ++kt) {
        bfx16 a = load_a_frag(o, D_, rowBase, kt * 32, lane);
#pragma unroll
        for (int j = 0; j < 4; ++j) {
            bfx16 b = load_b_frag(wop, NT, kt, wv_ * 4 + j, lane);
            acc[j] = wmma_bf16(a, b, acc[j]);
        }
    }
    int m0 = lane & 15, half = lane >> 4;
#pragma unroll
    for (int j = 0; j < 4; ++j) {
        int col = (wv_ * 4 + j) * 16 + m0;
        float bb = bo[col], g = gamma[col];
#pragma unroll
        for (int r = 0; r < 8; ++r) {
            int row = rowBase + r + 8 * half;
            size_t idx = (size_t)row * D_ + col;
            dout[idx] = x[idx] + g * (acc[j][r] + bb);
        }
    }
}

// ---------------- kernel 7: fused MLP (gelu(h2@w1+b1))@w2+b2, residual in d_out ----------------
__global__ __launch_bounds__(256)
void mlp_kernel(const unsigned short* __restrict__ h2,
                const unsigned short* __restrict__ w1p,
                const float* __restrict__ b1,
                const unsigned short* __restrict__ w2p,
                const float* __restrict__ b2,
                const float* __restrict__ gmlp,
                float* __restrict__ dout) {
    __shared__ unsigned short lds[16 * 1032];   // 16 rows x 1024 (+8 pad) bf16 hidden
    int rowBase = blockIdx.x * 16;
    int wv_ = threadIdx.x >> 5, lane = threadIdx.x & 31;
    int m0 = lane & 15, half = lane >> 4;

    // phase 1: hidden = gelu(h2 @ w1 + b1)   (each wave: 16 x 128 slice)
    v8f acc[8];
#pragma unroll
    for (int j = 0; j < 8; ++j) acc[j] = (v8f){0,0,0,0,0,0,0,0};
    const int NT1 = HID_ / 16;
#pragma unroll
    for (int kt = 0; kt < 8; ++kt) {
        bfx16 a = load_a_frag(h2, D_, rowBase, kt * 32, lane);
#pragma unroll
        for (int j = 0; j < 8; ++j) {
            bfx16 b = load_b_frag(w1p, NT1, kt, wv_ * 8 + j, lane);
            acc[j] = wmma_bf16(a, b, acc[j]);
        }
    }
#pragma unroll
    for (int j = 0; j < 8; ++j) {
        int col = (wv_ * 8 + j) * 16 + m0;
        float bb = b1[col];
#pragma unroll
        for (int r = 0; r < 8; ++r) {
            int lr = r + 8 * half;
            lds[lr * 1032 + col] = f2bf(gelu_t(acc[j][r] + bb));
        }
    }
    __syncthreads();

    // phase 2: out = x1 + gmlp * (hidden @ w2 + b2)   (each wave: 16 x 32 slice)
    v8f acc2[2];
#pragma unroll
    for (int j = 0; j < 2; ++j) acc2[j] = (v8f){0,0,0,0,0,0,0,0};
    const int NT2 = D_ / 16;
    for (int kt = 0; kt < 32; ++kt) {
        const unsigned short* p = lds + m0 * 1032 + kt * 32 + half * 8;
        Frag16 f;
        f.q[0] = *(const u32x4*)(p);
        f.q[1] = *(const u32x4*)(p + 16);
        bfx16 a = f.bf;
#pragma unroll
        for (int j = 0; j < 2; ++j) {
            bfx16 b = load_b_frag(w2p, NT2, kt, wv_ * 2 + j, lane);
            acc2[j] = wmma_bf16(a, b, acc2[j]);
        }
    }
#pragma unroll
    for (int j = 0; j < 2; ++j) {
        int col = (wv_ * 2 + j) * 16 + m0;
        float bb = b2[col], g = gmlp[col];
#pragma unroll
        for (int r = 0; r < 8; ++r) {
            int row = rowBase + r + 8 * half;
            size_t idx = (size_t)row * D_ + col;
            dout[idx] = dout[idx] + g * (acc2[j][r] + bb);
        }
    }
}

// ---------------- launch ----------------
extern "C" void kernel_launch(void* const* d_in, const int* in_sizes, int n_in,
                              void* d_out, int out_size, void* d_ws, size_t ws_size,
                              hipStream_t stream) {
    const float* x        = (const float*)d_in[0];
    const float* emb      = (const float*)d_in[1];
    const unsigned char* mask = (const unsigned char*)d_in[2];
    const float* w_mod1   = (const float*)d_in[3];
    const float* b_mod1   = (const float*)d_in[4];
    const float* wq       = (const float*)d_in[5];
    const float* bq       = (const float*)d_in[6];
    const float* wk       = (const float*)d_in[7];
    const float* bk       = (const float*)d_in[8];
    const float* wv       = (const float*)d_in[9];
    const float* bv       = (const float*)d_in[10];
    const float* wo       = (const float*)d_in[11];
    const float* bo       = (const float*)d_in[12];
    const float* gamma    = (const float*)d_in[13];
    const float* w_mod2   = (const float*)d_in[14];
    const float* b_mod2   = (const float*)d_in[15];
    const float* w1       = (const float*)d_in[16];
    const float* b1       = (const float*)d_in[17];
    const float* w2       = (const float*)d_in[18];
    const float* b2       = (const float*)d_in[19];
    const float* gmlp     = (const float*)d_in[20];
    float* dout = (float*)d_out;

    // workspace layout
    char* ws = (char*)d_ws;
    float* mod1 = (float*)ws;                 ws += (size_t)B_ * 2 * M_ * 4;
    float* mod2 = (float*)ws;                 ws += (size_t)B_ * 2 * M_ * 4;
    unsigned short* wqkv_p = (unsigned short*)ws; ws += (size_t)3 * D_ * D_ * 2;
    unsigned short* wo_p   = (unsigned short*)ws; ws += (size_t)D_ * D_ * 2;
    unsigned short* w1_p   = (unsigned short*)ws; ws += (size_t)D_ * HID_ * 2;
    unsigned short* w2_p   = (unsigned short*)ws; ws += (size_t)HID_ * D_ * 2;
    unsigned short* hbuf   = (unsigned short*)ws; ws += (size_t)R_ * D_ * 2;  // h / h2
    unsigned short* qkv    = (unsigned short*)ws; ws += (size_t)3 * R_ * D_ * 2;
    unsigned short* obuf   = (unsigned short*)ws; ws += (size_t)R_ * D_ * 2;

    // 1. conditioning projections
    mod_kernel<<<64, 256, 0, stream>>>(emb, w_mod1, b_mod1, w_mod2, b_mod2, mod1, mod2);

    // 2. weight packing (bf16 WMMA-B tile order)
    pack_kernel<<<(D_ * D_) / 256, 256, 0, stream>>>(wq, wqkv_p,                D_, D_);
    pack_kernel<<<(D_ * D_) / 256, 256, 0, stream>>>(wk, wqkv_p + D_ * D_,      D_, D_);
    pack_kernel<<<(D_ * D_) / 256, 256, 0, stream>>>(wv, wqkv_p + 2 * D_ * D_,  D_, D_);
    pack_kernel<<<(D_ * D_) / 256, 256, 0, stream>>>(wo, wo_p,                  D_, D_);
    pack_kernel<<<(D_ * HID_) / 256, 256, 0, stream>>>(w1, w1_p, D_, HID_);
    pack_kernel<<<(HID_ * D_) / 256, 256, 0, stream>>>(w2, w2_p, HID_, D_);

    // 3. adaLN #1 : x -> h (bf16)
    adaln_kernel<<<T_, 256, 0, stream>>>(x, mod1, hbuf);

    // 4. QKV projections (WMMA)
    gemm_qkv_kernel<<<dim3(R_ / 16, 3), 128, 0, stream>>>(hbuf, wqkv_p, bq, bk, bv, qkv);

    // 5. masked attention over variables
    attn_kernel<<<T_, 256, 0, stream>>>(qkv, qkv + (size_t)R_ * D_,
                                        qkv + (size_t)2 * R_ * D_, mask, obuf);

    // 6. out projection + gamma residual -> x1 in d_out (WMMA)
    gemm_oproj_kernel<<<R_ / 16, 128, 0, stream>>>(obuf, wo_p, bo, gamma, x, dout);

    // 7. adaLN #2 : x1 -> h2 (reuse hbuf)
    adaln_kernel<<<T_, 256, 0, stream>>>(dout, mod2, hbuf);

    // 8. fused MLP + gamma_mlp residual (WMMA, hidden via LDS)
    mlp_kernel<<<R_ / 16, 256, 0, stream>>>(hbuf, w1_p, b1, w2_p, b2, gmlp, dout);
}
